// HapbertaAxialEncoder_2903397892817
// MI455X (gfx1250) — matrix operand order repaired and checked by reading, not compiled, not run on hardware
//
#include <hip/hip_runtime.h>
#include <hip/hip_bf16.h>
#include <math.h>

typedef __attribute__((ext_vector_type(16))) unsigned short v16u;
typedef __attribute__((ext_vector_type(16))) __bf16        v16bf;
typedef __attribute__((ext_vector_type(8)))  float         v8f;

#define DEVINL __device__ __forceinline__

constexpr int cB = 2, cR = 32, cC = 256, cH = 256, cNH = 8, cHD = 32, cI = 1024, cL = 2, cNB = 32;
constexpr int TOK = cB * cR * cC;  // 16384 tokens, layout [B,R,C,H] throughout

// ---------- bf16 helpers (bit-level, no __bf16 arithmetic) ----------
DEVINL unsigned short f2b(float f) {
  unsigned u = __builtin_bit_cast(unsigned, f);
  unsigned r = (u + 0x7FFFu + ((u >> 16) & 1u)) >> 16;  // RNE
  return (unsigned short)r;
}
__attribute__((unused)) DEVINL float b2f(unsigned short h) {
  unsigned u = ((unsigned)h) << 16;
  return __builtin_bit_cast(float, u);
}

// WMMA bf16 16x16x32 fragment K-index maps (per CDNA5 ISA §7.12.2)
// A (16x32): lanes 0-15 hold K {0..7,16..23}, lanes 16-31 hold K {8..15,24..31}
DEVINL int kA(int v, int half) { return (v < 4 ? 2 * v : 16 + 2 * (v - 4)) + 8 * half; }
// B (32x16): lanes 0-15 hold K {0..7,16..23}, lanes 16-31 hold K {8..15,24..31}
DEVINL int kB(int v, int half) { return (v < 4 ? 2 * v : 16 + 2 * (v - 4)) + 8 * half; }

DEVINL v8f wmma_bf16(v16u a, v16u b, v8f c) {
  return __builtin_amdgcn_wmma_f32_16x16x32_bf16(
      false, __builtin_bit_cast(v16bf, a),
      false, __builtin_bit_cast(v16bf, b),
      (short)0, c, false, false);
}

// ---------- CDNA5 async global->LDS copy (ASYNCcnt path, ISA §15.18.3 op 98) ----------
// LDS aperture occupies addr[63:32]; addr[31:0] is the LDS byte address (ISA §10.2),
// so the low 32 bits of a generic __shared__ pointer are a valid VDST operand.
DEVINL void async_ld_b128(void* lds_ptr, const void* gptr) {
  unsigned lds_off = (unsigned)(size_t)lds_ptr;
  asm volatile("global_load_async_to_lds_b128 %0, %1, off"
               :: "v"(lds_off), "v"(gptr)
               : "memory");
}
DEVINL void wait_async0() { asm volatile("s_wait_asynccnt 0" ::: "memory"); }

// ---------- f32 -> bf16 weight conversion ----------
__global__ void cvt_bf16_kernel(const float* __restrict__ src, unsigned short* __restrict__ dst, int n) {
  int i = blockIdx.x * 256 + threadIdx.x;
  if (i < n) dst[i] = f2b(src[i]);
}

// ---------- LayerNorm over H=256, one wave per token ----------
template <bool BF16OUT>
__global__ __launch_bounds__(256) void ln_kernel(const float* __restrict__ x,
                                                 const float* __restrict__ w,
                                                 const float* __restrict__ b,
                                                 void* __restrict__ out) {
  const int wid = threadIdx.x >> 5, lane = threadIdx.x & 31;
  const int tok = blockIdx.x * 8 + wid;
  const float* row = x + (size_t)tok * cH;
  float v[8];
  float s = 0.f;
#pragma unroll
  for (int e = 0; e < 8; e++) { v[e] = row[e * 32 + lane]; s += v[e]; }
#pragma unroll
  for (int m = 16; m; m >>= 1) s += __shfl_xor(s, m, 32);
  const float mean = s * (1.0f / cH);
  float vs = 0.f;
#pragma unroll
  for (int e = 0; e < 8; e++) { float d = v[e] - mean; vs += d * d; }
#pragma unroll
  for (int m = 16; m; m >>= 1) vs += __shfl_xor(vs, m, 32);
  const float inv = rsqrtf(vs * (1.0f / cH) + 1e-12f);
#pragma unroll
  for (int e = 0; e < 8; e++) {
    const int ci = e * 32 + lane;
    const float y = (v[e] - mean) * inv * w[ci] + b[ci];
    if (BF16OUT)
      ((unsigned short*)out)[(size_t)tok * cH + ci] = f2b(y);
    else
      ((float*)out)[(size_t)tok * cH + ci] = y;
  }
}

// ---------- GEMM: out[M,N] = A[M,K](bf16) x W[N,K]^T(bf16) + bias, fused epilogues ----------
// Double-buffered async-to-LDS pipeline: next 32-wide K slab streams in (ASYNCcnt)
// while the current slab feeds v_wmma; one barrier per K-step.
// EPI 0: bf16 out, * scale   EPI 1: bf16 out, exact GELU   EPI 2: f32 out, + residual
template <int EPI>
__global__ __launch_bounds__(256) void gemm_kernel(const unsigned short* __restrict__ A,
                                                   const unsigned short* __restrict__ Bw,
                                                   const float* __restrict__ bias,
                                                   const float* __restrict__ resid,
                                                   float scale,
                                                   float* __restrict__ outF,
                                                   unsigned short* __restrict__ outB,
                                                   int M, int N, int K) {
  __shared__ unsigned short As[2][128][32];  // [buf][m][k]
  __shared__ unsigned short Bs[2][128][32];  // [buf][n][k] (K-contiguous: async-copy friendly)
  const int t = threadIdx.x;
  const int wid = t >> 5, lane = t & 31, half = lane >> 4, mr = lane & 15;
  const int wm = wid & 1, wn = wid >> 1;  // 2 x 4 wave grid
  const int m0 = blockIdx.x * 128, n0 = blockIdx.y * 128;
  v8f zero = {};
  v8f acc[4][2];
#pragma unroll
  for (int i = 0; i < 4; i++)
#pragma unroll
    for (int j = 0; j < 2; j++) acc[i][j] = zero;

  // stage one 128x32 A slab + 128x32 B slab: 2 x 16B async loads per thread each
  auto stage = [&](int buf, int k0) {
#pragma unroll
    for (int e = 0; e < 2; e++) {
      const int idx = t + e * 256;   // 0..511 chunks of 8 halves
      const int rw = idx >> 2;       // 0..127
      const int c8 = (idx & 3) * 8;  // half offset within the 32-wide slab
      async_ld_b128(&As[buf][rw][c8], &A[(size_t)(m0 + rw) * K + k0 + c8]);
      async_ld_b128(&Bs[buf][rw][c8], &Bw[(size_t)(n0 + rw) * K + k0 + c8]);
    }
  };

  stage(0, 0);
  const int steps = K >> 5;
  for (int s = 0; s < steps; s++) {
    const int cur = s & 1;
    wait_async0();       // this wave's slab arrived
    __syncthreads();     // all waves' slab arrived; prev buffer fully consumed
    if (s + 1 < steps) stage(cur ^ 1, (s + 1) * 32);

    v16u af[4], bfr[2];
#pragma unroll
    for (int tm = 0; tm < 4; tm++) {
      const int rowA = wm * 64 + tm * 16 + mr;
#pragma unroll
      for (int v = 0; v < 8; v++) {
        const int k = kA(v, half);
        af[tm][2 * v] = As[cur][rowA][k];
        af[tm][2 * v + 1] = As[cur][rowA][k + 1];
      }
    }
#pragma unroll
    for (int tn = 0; tn < 2; tn++) {
      const int colB = wn * 32 + tn * 16 + mr;
#pragma unroll
      for (int v = 0; v < 8; v++) {
        const int k = kB(v, half);
        bfr[tn][2 * v] = Bs[cur][colB][k];
        bfr[tn][2 * v + 1] = Bs[cur][colB][k + 1];
      }
    }
#pragma unroll
    for (int tm = 0; tm < 4; tm++)
#pragma unroll
      for (int tn = 0; tn < 2; tn++) acc[tm][tn] = wmma_bf16(af[tm], bfr[tn], acc[tm][tn]);
  }

#pragma unroll
  for (int tm = 0; tm < 4; tm++)
#pragma unroll
    for (int tn = 0; tn < 2; tn++) {
      const int col = n0 + wn * 32 + tn * 16 + mr;
      const float bv = bias[col];
#pragma unroll
      for (int r = 0; r < 8; r++) {
        const int row = m0 + wm * 64 + tm * 16 + 8 * half + r;
        const float val = acc[tm][tn][r] + bv;
        const size_t off = (size_t)row * N + col;
        if (EPI == 0) {
          outB[off] = f2b(val * scale);
        } else if (EPI == 1) {
          const float g = 0.5f * val * (1.0f + erff(val * 0.70710678118f));
          outB[off] = f2b(g);
        } else {
          outF[off] = val + resid[off];
        }
      }
    }
}

// ---------- Row attention: WG = (b*R+r, head, 32-row block), softmax over C=256 ----------
__global__ __launch_bounds__(256) void row_attn_kernel(const unsigned short* __restrict__ Qb,
                                                       const unsigned short* __restrict__ Kb,
                                                       const unsigned short* __restrict__ Vb,
                                                       const int* __restrict__ dist,
                                                       const float* __restrict__ rel,
                                                       unsigned short* __restrict__ ctx) {
  __shared__ float S[32][264];           // 32 query rows x 256 keys (+pad)
  __shared__ unsigned short Vt[256][32];
  __shared__ float red[32][8];
  const int t = threadIdx.x;
  const int wid = t >> 5, lane = t & 31, half = lane >> 4, mr = lane & 15;
  const int br = blockIdx.x;  // b*R + r
  const int h = blockIdx.y;
  const int i0 = blockIdx.z * 32;
  const size_t tb = (size_t)br * cC;  // base token (b,r,0)

  // stage V[j][d] for this head via async copies (consumed after the barriers below)
#pragma unroll
  for (int d = 0; d < 32; d += 8)
    async_ld_b128(&Vt[t][d], &Vb[(tb + t) * cH + h * cHD + d]);

  // Q fragments (kept in registers across all key tiles); q already scaled in GEMM
  v16u aq[2];
#pragma unroll
  for (int tm = 0; tm < 2; tm++) {
    const unsigned short* qr = &Qb[(tb + i0 + tm * 16 + mr) * cH + h * cHD];
#pragma unroll
    for (int v = 0; v < 8; v++) {
      const int k = kA(v, half);
      aq[tm][2 * v] = qr[k];
      aq[tm][2 * v + 1] = qr[k + 1];
    }
  }
  // each wave owns 32 key columns; HD=32 -> single WMMA K-step per tile
  const int j0 = wid * 32;
  v16u bk2[2];
#pragma unroll
  for (int tn = 0; tn < 2; tn++) {
    const unsigned short* kr = &Kb[(tb + j0 + tn * 16 + mr) * cH + h * cHD];
#pragma unroll
    for (int v = 0; v < 8; v++) {
      const int k = kB(v, half);
      bk2[tn][2 * v] = kr[k];
      bk2[tn][2 * v + 1] = kr[k + 1];
    }
  }
  v8f zero = {};
#pragma unroll
  for (int tm = 0; tm < 2; tm++)
#pragma unroll
    for (int tn = 0; tn < 2; tn++) {
      v8f s = wmma_bf16(aq[tm], bk2[tn], zero);
#pragma unroll
      for (int r = 0; r < 8; r++) S[tm * 16 + 8 * half + r][j0 + tn * 16 + mr] = s[r];
    }
  wait_async0();  // V slab resident before first block-wide barrier
  __syncthreads();

  // softmax with T5-style log-bucket relative bias (8 threads per row)
  const int row = t >> 3, seg = t & 7;
  const int* drow = &dist[((size_t)br * cC + (i0 + row)) * cC];
  float mx = -3.0e38f;
  for (int j = seg * 32; j < seg * 32 + 32; j++) {
    const int d = drow[j];
    const float a = fabsf((float)d);
    int bkt;
    if (a < 1.0f) {
      bkt = 0;
    } else {
      int lb = (int)(logf(a) * (30.0f / 10.819778284410283f)) + 1;  // (NB-2)/log(50000)
      bkt = lb < 1 ? 1 : (lb > 31 ? 31 : lb);
    }
    const float sv = S[row][j] + rel[bkt * cNH + h];
    S[row][j] = sv;
    mx = fmaxf(mx, sv);
  }
  red[row][seg] = mx;
  __syncthreads();
  mx = red[row][0];
#pragma unroll
  for (int s2 = 1; s2 < 8; s2++) mx = fmaxf(mx, red[row][s2]);
  __syncthreads();
  float sum = 0.f;
  for (int j = seg * 32; j < seg * 32 + 32; j++) {
    const float e = __expf(S[row][j] - mx);
    S[row][j] = e;
    sum += e;
  }
  red[row][seg] = sum;
  __syncthreads();
  sum = 0.f;
#pragma unroll
  for (int s2 = 0; s2 < 8; s2++) sum += red[row][s2];
  const float inv = 1.0f / sum;
  for (int j = seg * 32; j < seg * 32 + 32; j++) S[row][j] *= inv;
  __syncthreads();

  // O = P(32x256) x V(256x32); 4 output tiles on waves 0..3, K = 8 WMMA steps
  if (wid < 4) {
    const int tm = wid & 1, tn = wid >> 1;
    v8f o = {};
    for (int k0 = 0; k0 < 256; k0 += 32) {
      v16u pa, vb2;
#pragma unroll
      for (int v = 0; v < 8; v++) {
        const int ka = kA(v, half);
        pa[2 * v] = f2b(S[tm * 16 + mr][k0 + ka]);
        pa[2 * v + 1] = f2b(S[tm * 16 + mr][k0 + ka + 1]);
        const int kb2 = kB(v, half);
        vb2[2 * v] = Vt[k0 + kb2][tn * 16 + mr];
        vb2[2 * v + 1] = Vt[k0 + kb2 + 1][tn * 16 + mr];
      }
      o = wmma_bf16(pa, vb2, o);
    }
#pragma unroll
    for (int r = 0; r < 8; r++)
      ctx[(tb + i0 + tm * 16 + 8 * half + r) * cH + h * cHD + tn * 16 + mr] = f2b(o[r]);
  }
}

// ---------- Col attention: WG = (b, c), wave = head, softmax over R=32 ----------
__global__ __launch_bounds__(256) void col_attn_kernel(const unsigned short* __restrict__ Qb,
                                                       const unsigned short* __restrict__ Kb,
                                                       const unsigned short* __restrict__ Vb,
                                                       unsigned short* __restrict__ ctx) {
  __shared__ float S[cNH][32][33];
  const int t = threadIdx.x;
  const int h = t >> 5, lane = t & 31, half = lane >> 4, mr = lane & 15;
  const int bc = blockIdx.x;
  const int b = bc >> 8, c = bc & 255;
  const size_t base = ((size_t)b * cR) * cC + c;  // token(j) = base + j*cC

  v16u aq[2], bk2[2];
#pragma unroll
  for (int tm = 0; tm < 2; tm++) {
    const unsigned short* qr = &Qb[(base + (size_t)(tm * 16 + mr) * cC) * cH + h * cHD];
#pragma unroll
    for (int v = 0; v < 8; v++) {
      const int k = kA(v, half);
      aq[tm][2 * v] = qr[k];
      aq[tm][2 * v + 1] = qr[k + 1];
    }
  }
#pragma unroll
  for (int tn = 0; tn < 2; tn++) {
    const unsigned short* kr = &Kb[(base + (size_t)(tn * 16 + mr) * cC) * cH + h * cHD];
#pragma unroll
    for (int v = 0; v < 8; v++) {
      const int k = kB(v, half);
      bk2[tn][2 * v] = kr[k];
      bk2[tn][2 * v + 1] = kr[k + 1];
    }
  }
  v8f zero = {};
#pragma unroll
  for (int tm = 0; tm < 2; tm++)
#pragma unroll
    for (int tn = 0; tn < 2; tn++) {
      v8f s = wmma_bf16(aq[tm], bk2[tn], zero);
#pragma unroll
      for (int r = 0; r < 8; r++) S[h][tm * 16 + 8 * half + r][tn * 16 + mr] = s[r];
    }
  __syncthreads();

  {  // per-lane softmax, row = lane (wave32 -> 32 rows)
    float mx = -3.0e38f;
#pragma unroll
    for (int j = 0; j < 32; j++) mx = fmaxf(mx, S[h][lane][j]);
    float sum = 0.f;
#pragma unroll
    for (int j = 0; j < 32; j++) {
      const float e = __expf(S[h][lane][j] - mx);
      S[h][lane][j] = e;
      sum += e;
    }
    const float inv = 1.0f / sum;
#pragma unroll
    for (int j = 0; j < 32; j++) S[h][lane][j] *= inv;
  }
  __syncthreads();

  // O = P(32x32) x V(32x32): one WMMA per output tile (K = R = 32)
#pragma unroll
  for (int tm = 0; tm < 2; tm++)
#pragma unroll
    for (int tn = 0; tn < 2; tn++) {
      v16u pa, vb2;
#pragma unroll
      for (int v = 0; v < 8; v++) {
        const int ka = kA(v, half);
        pa[2 * v] = f2b(S[h][tm * 16 + mr][ka]);
        pa[2 * v + 1] = f2b(S[h][tm * 16 + mr][ka + 1]);
        const int kb3 = kB(v, half);
        vb2[2 * v] = Vb[(base + (size_t)kb3 * cC) * cH + h * cHD + tn * 16 + mr];
        vb2[2 * v + 1] = Vb[(base + (size_t)(kb3 + 1) * cC) * cH + h * cHD + tn * 16 + mr];
      }
      v8f o = wmma_bf16(pa, vb2, zero);
#pragma unroll
      for (int r = 0; r < 8; r++)
        ctx[(base + (size_t)(tm * 16 + 8 * half + r) * cC) * cH + h * cHD + tn * 16 + mr] = f2b(o[r]);
    }
}

// ---------------------------------------------------------------------------
extern "C" void kernel_launch(void* const* d_in, const int* in_sizes, int n_in,
                              void* d_out, int out_size, void* d_ws, size_t ws_size,
                              hipStream_t stream) {
  (void)in_sizes; (void)n_in; (void)out_size; (void)ws_size;

  const float* p_hs     = (const float*)d_in[0];
  const int*   p_dist   = (const int*)d_in[1];
  const float* p_rqW    = (const float*)d_in[2];
  const float* p_rqb    = (const float*)d_in[3];
  const float* p_rkW    = (const float*)d_in[4];
  const float* p_rkb    = (const float*)d_in[5];
  const float* p_rvW    = (const float*)d_in[6];
  const float* p_rvb    = (const float*)d_in[7];
  const float* p_roW    = (const float*)d_in[8];
  const float* p_rob    = (const float*)d_in[9];
  const float* p_rrel   = (const float*)d_in[10];
  const float* p_cqW    = (const float*)d_in[11];
  const float* p_cqb    = (const float*)d_in[12];
  const float* p_ckW    = (const float*)d_in[13];
  const float* p_ckb    = (const float*)d_in[14];
  const float* p_cvW    = (const float*)d_in[15];
  const float* p_cvb    = (const float*)d_in[16];
  const float* p_coW    = (const float*)d_in[17];
  const float* p_cob    = (const float*)d_in[18];
  const float* p_f1W    = (const float*)d_in[19];
  const float* p_f1b    = (const float*)d_in[20];
  const float* p_f2W    = (const float*)d_in[21];
  const float* p_f2b    = (const float*)d_in[22];
  const float* p_lnrw   = (const float*)d_in[23];
  const float* p_lnrb   = (const float*)d_in[24];
  const float* p_lncw   = (const float*)d_in[25];
  const float* p_lncb   = (const float*)d_in[26];
  const float* p_lnfw   = (const float*)d_in[27];
  const float* p_lnfb   = (const float*)d_in[28];
  const float* p_lnFw   = (const float*)d_in[29];
  const float* p_lnFb   = (const float*)d_in[30];

  // workspace layout (~63 MB)
  char* p = (char*)d_ws;
  float* x = (float*)p;                 p += (size_t)TOK * cH * 4;    // hidden state f32
  unsigned short* xn = (unsigned short*)p; p += (size_t)TOK * cH * 2; // LN output bf16
  unsigned short* qb = (unsigned short*)p;                            // q/k/v/ctx bf16
  unsigned short* kb2 = qb + (size_t)TOK * cH;
  unsigned short* vb = kb2 + (size_t)TOK * cH;
  unsigned short* cx = vb + (size_t)TOK * cH;
  unsigned short* h1 = qb;  // FFN hidden aliases q/k/v/ctx region (TOK x 1024 bf16)
  p += (size_t)TOK * cI * 2;
  unsigned short* wp = (unsigned short*)p;

  auto conv = [&](const float* src, int n) -> unsigned short* {
    unsigned short* d = wp;
    wp += n;
    cvt_bf16_kernel<<<(n + 255) / 256, 256, 0, stream>>>(src, d, n);
    return d;
  };

  unsigned short *wrq[cL], *wrk[cL], *wrv[cL], *wro[cL];
  unsigned short *wcq[cL], *wck[cL], *wcv[cL], *wco[cL];
  unsigned short *wf1[cL], *wf2[cL];
  for (int l = 0; l < cL; l++) {
    wrq[l] = conv(p_rqW + (size_t)l * cH * cH, cH * cH);
    wrk[l] = conv(p_rkW + (size_t)l * cH * cH, cH * cH);
    wrv[l] = conv(p_rvW + (size_t)l * cH * cH, cH * cH);
    wro[l] = conv(p_roW + (size_t)l * cH * cH, cH * cH);
    wcq[l] = conv(p_cqW + (size_t)l * cH * cH, cH * cH);
    wck[l] = conv(p_ckW + (size_t)l * cH * cH, cH * cH);
    wcv[l] = conv(p_cvW + (size_t)l * cH * cH, cH * cH);
    wco[l] = conv(p_coW + (size_t)l * cH * cH, cH * cH);
    wf1[l] = conv(p_f1W + (size_t)l * cI * cH, cI * cH);
    wf2[l] = conv(p_f2W + (size_t)l * cH * cI, cH * cI);
  }

  hipMemcpyAsync(x, p_hs, (size_t)TOK * cH * 4, hipMemcpyDeviceToDevice, stream);

  const float scale_row = 0.03125f;              // HD^-0.5 / sqrt(R) = 1/32
  const float scale_col = 0.17677669529663687f;  // HD^-0.5
  const dim3 g256(128, 2), g1024(128, 8);
  const int lnGrid = TOK / 8;

  for (int l = 0; l < cL; l++) {
    // --- row attention block ---
    ln_kernel<true><<<lnGrid, 256, 0, stream>>>(x, p_lnrw + l * cH, p_lnrb + l * cH, xn);
    gemm_kernel<0><<<g256, 256, 0, stream>>>(xn, wrq[l], p_rqb + l * cH, nullptr, scale_row, nullptr, qb, TOK, cH, cH);
    gemm_kernel<0><<<g256, 256, 0, stream>>>(xn, wrk[l], p_rkb + l * cH, nullptr, 1.0f, nullptr, kb2, TOK, cH, cH);
    gemm_kernel<0><<<g256, 256, 0, stream>>>(xn, wrv[l], p_rvb + l * cH, nullptr, 1.0f, nullptr, vb, TOK, cH, cH);
    row_attn_kernel<<<dim3(cB * cR, cNH, cC / 32), 256, 0, stream>>>(
        qb, kb2, vb, p_dist, p_rrel + (size_t)l * cNB * cNH, cx);
    gemm_kernel<2><<<g256, 256, 0, stream>>>(cx, wro[l], p_rob + l * cH, x, 1.0f, x, nullptr, TOK, cH, cH);

    // --- col attention block ---
    ln_kernel<true><<<lnGrid, 256, 0, stream>>>(x, p_lncw + l * cH, p_lncb + l * cH, xn);
    gemm_kernel<0><<<g256, 256, 0, stream>>>(xn, wcq[l], p_cqb + l * cH, nullptr, scale_col, nullptr, qb, TOK, cH, cH);
    gemm_kernel<0><<<g256, 256, 0, stream>>>(xn, wck[l], p_ckb + l * cH, nullptr, 1.0f, nullptr, kb2, TOK, cH, cH);
    gemm_kernel<0><<<g256, 256, 0, stream>>>(xn, wcv[l], p_cvb + l * cH, nullptr, 1.0f, nullptr, vb, TOK, cH, cH);
    col_attn_kernel<<<cB * cC, 256, 0, stream>>>(qb, kb2, vb, cx);
    gemm_kernel<2><<<g256, 256, 0, stream>>>(cx, wco[l], p_cob + l * cH, x, 1.0f, x, nullptr, TOK, cH, cH);

    // --- FFN ---
    ln_kernel<true><<<lnGrid, 256, 0, stream>>>(x, p_lnfw + l * cH, p_lnfb + l * cH, xn);
    gemm_kernel<1><<<g1024, 256, 0, stream>>>(xn, wf1[l], p_f1b + l * cI, nullptr, 1.0f, nullptr, h1, TOK, cI, cH);
    gemm_kernel<2><<<g256, 256, 0, stream>>>(h1, wf2[l], p_f2b + l * cH, x, 1.0f, x, nullptr, TOK, cH, cI);
  }

  ln_kernel<false><<<lnGrid, 256, 0, stream>>>(x, p_lnFw, p_lnFb, d_out);
}
